// LSTM_84559316124385
// MI455X (gfx1250) — matrix operand order repaired and checked by reading, not compiled, not run on hardware
//
#include <hip/hip_runtime.h>
#include <hip/hip_bf16.h>

// Problem constants (match reference)
#define B_ 64
#define T_ 512
#define D_ 512
#define H_ 512
#define G_ 2048   // 4*H

typedef __attribute__((ext_vector_type(16))) __bf16 bf16x16;
typedef __attribute__((ext_vector_type(8)))  __bf16 bf16x8;
typedef __attribute__((ext_vector_type(8)))  float  f32x8;

// ---------------------------------------------------------------------------
// WMMA fragment loaders (this code's contiguous k = 16h + s order for A and B, wave32)
// A (16x32 bf16): lanes 0-15 -> M=lane, K=0..15;
//                 lanes 16-31 -> M=lane-16, K=16..31
// ---------------------------------------------------------------------------
static __device__ __forceinline__ bf16x16 load_a_frag(const __bf16* __restrict__ base, int ldk) {
    const int lane = threadIdx.x & 31;
    const int row  = lane & 15;
    const int hi   = lane >> 4;
    const __bf16* r = base + (size_t)row * ldk;
    return *(const bf16x16*)(r + hi * 16);
}

// B (32x16 bf16) from row-major W[N,K] (B = W^T tile):
// lane%16 = output column n (a row of W); lanes 0-15 hold K=0..15, lanes 16-31 K=16..31
static __device__ __forceinline__ bf16x16 load_b_frag(const __bf16* __restrict__ base, int ldk) {
    const int lane = threadIdx.x & 31;
    const int col  = lane & 15;
    const int hi   = lane >> 4;
    return *(const bf16x16*)(base + (size_t)col * ldk + hi * 16);
}

static __device__ __forceinline__ f32x8 wmma_bf16(bf16x16 a, bf16x16 b, f32x8 c) {
    return __builtin_amdgcn_wmma_f32_16x16x32_bf16(false, a, false, b, (short)0, c, false, false);
}

static __device__ __forceinline__ float sigmoidf_(float x) {
    return 1.0f / (1.0f + __expf(-x));
}

// ---------------------------------------------------------------------------
// CDNA5 async global->LDS copy (ASYNCcnt-tracked). LDS operand is the low 32
// bits of the generic pointer (LDS aperture truncation, ISA 10.2).
// ---------------------------------------------------------------------------
static __device__ __forceinline__ void async_load_b128(const void* gaddr, void* lds) {
    unsigned int lds_off = (unsigned int)(unsigned long long)lds;
    unsigned long long ga = (unsigned long long)gaddr;
    asm volatile("global_load_async_to_lds_b128 %0, %1, off"
                 :: "v"(lds_off), "v"(ga) : "memory");
}
static __device__ __forceinline__ void wait_asynccnt0() {
    asm volatile("s_wait_asynccnt 0" ::: "memory");
}

// ---------------------------------------------------------------------------
// fp32 -> bf16 conversion (grid-stride)
// ---------------------------------------------------------------------------
__global__ __launch_bounds__(256) void cvt_f32_bf16(const float* __restrict__ in,
                                                    __bf16* __restrict__ out, int n) {
    for (int i = blockIdx.x * 256 + threadIdx.x; i < n; i += gridDim.x * 256)
        out[i] = (__bf16)in[i];
}

__global__ __launch_bounds__(256) void bias_sum(const float* __restrict__ a,
                                                const float* __restrict__ b,
                                                float* __restrict__ out, int n) {
    int i = blockIdx.x * 256 + threadIdx.x;
    if (i < n) out[i] = a[i] + b[i];
}

// ---------------------------------------------------------------------------
// Batched GEMM:  Dst[M,N] = A[M,K] @ W[N,K]^T   (bf16 in, fp32 out)
// Block = 256 thr (8 waves, 4x2), tile 128x64, each wave 32x32 (4 WMMA tiles)
// ---------------------------------------------------------------------------
__global__ __launch_bounds__(256) void gemm_bt(const __bf16* __restrict__ A,
                                               const __bf16* __restrict__ W,
                                               float* __restrict__ Dst,
                                               int M, int N, int K) {
    const int wave = threadIdx.x >> 5;
    const int wm = wave & 3, wn = wave >> 2;
    const int m0 = blockIdx.x * 128 + wm * 32;
    const int n0 = blockIdx.y * 64 + wn * 32;

    f32x8 acc00 = {}, acc01 = {}, acc10 = {}, acc11 = {};
    const __bf16* a0p = A + (size_t)m0 * K;
    const __bf16* a1p = A + (size_t)(m0 + 16) * K;
    const __bf16* b0p = W + (size_t)n0 * K;
    const __bf16* b1p = W + (size_t)(n0 + 16) * K;

    for (int k = 0; k < K; k += 32) {
        __builtin_prefetch(a0p + k + 256, 0, 1);
        bf16x16 a0 = load_a_frag(a0p + k, K);
        bf16x16 a1 = load_a_frag(a1p + k, K);
        bf16x16 b0 = load_b_frag(b0p + k, K);
        bf16x16 b1 = load_b_frag(b1p + k, K);
        acc00 = wmma_bf16(a0, b0, acc00);
        acc01 = wmma_bf16(a0, b1, acc01);
        acc10 = wmma_bf16(a1, b0, acc10);
        acc11 = wmma_bf16(a1, b1, acc11);
    }

    const int lane = threadIdx.x & 31;
    const int ln = lane & 15, lh = lane >> 4;
#pragma unroll
    for (int v = 0; v < 8; ++v) {
        int r0 = m0 + v + 8 * lh, r1 = r0 + 16;
        Dst[(size_t)r0 * N + n0 + ln]      = acc00[v];
        Dst[(size_t)r0 * N + n0 + 16 + ln] = acc01[v];
        Dst[(size_t)r1 * N + n0 + ln]      = acc10[v];
        Dst[(size_t)r1 * N + n0 + 16 + ln] = acc11[v];
    }
}

// ---------------------------------------------------------------------------
// Persistent LSTM recurrence. Grid = 32 blocks; block b owns hidden units
// [16b, 16b+16) across all 4 gates. Per step:
//   1) issue async global->LDS loads of this timestep's xg slice (HBM stream,
//      latency hidden behind the WMMA chain)
//   2) WMMA [64,512]x[512,64] from the L2-resident h ping-pong buffer
//   3) s_wait_asynccnt 0, LSTM elementwise in LDS
//   4) grid barrier (atomic count + threadfence)
// hbuf: two [B_,H_] bf16 buffers (ping-pong), buf0 zeroed before launch.
// ---------------------------------------------------------------------------
__global__ __launch_bounds__(256) void lstm_scan(const float* __restrict__ xg,   // [B*T, 4H]
                                                 const __bf16* __restrict__ Whh, // [4H, H] bf16
                                                 const float* __restrict__ bias, // [4H] (b_ih+b_hh)
                                                 __bf16* __restrict__ hbuf,      // 2 x [B,H]
                                                 __bf16* __restrict__ hout,      // [B*T,H] or null
                                                 float* __restrict__ hlast,      // [B,H] or null
                                                 unsigned int* __restrict__ bar) {
    const int blk  = blockIdx.x;        // 0..31
    const int tid  = threadIdx.x;
    const int lane = tid & 31;
    const int wave = tid >> 5;
    const int mt   = wave & 3;          // batch tile (rows mt*16..+16)
    const int gp   = wave >> 2;         // 0 -> gates {0,1}, 1 -> gates {2,3}
    const int u0   = blk * 16;          // first hidden unit owned by this block
    const int m0   = mt * 16;
    const int ln = lane & 15, lh = lane >> 4;

    __shared__ float gbuf[64 * 64];     // [batch][gate*16 + u]   (h@Whh^T result)
    __shared__ float xstage[64 * 64];   // [batch][gate*16 + u]   (xg slice for t)
    __shared__ float cbuf[64 * 16];     // cell state for owned units
    for (int i = tid; i < 64 * 16; i += 256) cbuf[i] = 0.0f;
    __syncthreads();

    const __bf16* wb0 = Whh + (size_t)((gp * 2 + 0) * H_ + u0) * H_;
    const __bf16* wb1 = Whh + (size_t)((gp * 2 + 1) * H_ + u0) * H_;

    for (int t = 0; t < T_; ++t) {
        const __bf16* hread  = hbuf + (size_t)(t & 1) * (B_ * H_);
        __bf16*       hwrite = hbuf + (size_t)((t + 1) & 1) * (B_ * H_);

        // 1) async-stage xg[:, t, owned 64 gate cols] -> LDS (1024 x b128)
#pragma unroll
        for (int c = tid; c < 1024; c += 256) {
            int b = c >> 4, g = (c >> 2) & 3, q = c & 3;
            const float* src = xg + ((size_t)b * T_ + t) * G_ + g * H_ + u0 + q * 4;
            async_load_b128(src, &xstage[b * 64 + g * 16 + q * 4]);
        }

        // 2) recurrent GEMM on matrix cores
        f32x8 acc0 = {}, acc1 = {};
        const __bf16* ap = hread + (size_t)m0 * H_;
#pragma unroll 4
        for (int k = 0; k < H_; k += 32) {
            bf16x16 a  = load_a_frag(ap + k, H_);
            bf16x16 b0 = load_b_frag(wb0 + k, H_);
            bf16x16 b1 = load_b_frag(wb1 + k, H_);
            acc0 = wmma_bf16(a, b0, acc0);
            acc1 = wmma_bf16(a, b1, acc1);
        }
#pragma unroll
        for (int v = 0; v < 8; ++v) {
            int r = m0 + v + 8 * lh;
            gbuf[r * 64 + (gp * 2 + 0) * 16 + ln] = acc0[v];
            gbuf[r * 64 + (gp * 2 + 1) * 16 + ln] = acc1[v];
        }

        // 3) xg slice is needed now; wait for this wave's async copies, then
        //    barrier so every wave's LDS writes (gbuf + xstage) are visible.
        wait_asynccnt0();
        __syncthreads();

        // LSTM elementwise for owned units: 64 batches x 16 units
        for (int p = tid; p < 64 * 16; p += 256) {
            int b = p >> 4, u = p & 15;
            int cu = u0 + u;
            float gi = gbuf[b * 64 + 0 * 16 + u] + xstage[b * 64 + 0 * 16 + u] + bias[0 * H_ + cu];
            float gf = gbuf[b * 64 + 1 * 16 + u] + xstage[b * 64 + 1 * 16 + u] + bias[1 * H_ + cu];
            float gg = gbuf[b * 64 + 2 * 16 + u] + xstage[b * 64 + 2 * 16 + u] + bias[2 * H_ + cu];
            float go = gbuf[b * 64 + 3 * 16 + u] + xstage[b * 64 + 3 * 16 + u] + bias[3 * H_ + cu];
            gi = sigmoidf_(gi);
            gf = sigmoidf_(gf);
            go = sigmoidf_(go);
            gg = tanhf(gg);
            float c = gf * cbuf[p] + gi * gg;
            cbuf[p] = c;
            float h = go * tanhf(c);
            __bf16 hb = (__bf16)h;
            hwrite[(size_t)b * H_ + cu] = hb;
            if (hout)  hout[((size_t)b * T_ + t) * H_ + cu] = hb;
            if (hlast && t == T_ - 1) hlast[(size_t)b * H_ + cu] = h;
        }

        // 4) Grid-wide barrier: make h visible, then count arrivals.
        __threadfence();
        __syncthreads();
        if (tid == 0) {
            __hip_atomic_fetch_add(bar, 1u, __ATOMIC_RELEASE, __HIP_MEMORY_SCOPE_AGENT);
            unsigned int target = 32u * (unsigned int)(t + 1);
            while (__hip_atomic_load(bar, __ATOMIC_ACQUIRE, __HIP_MEMORY_SCOPE_AGENT) < target)
                __builtin_amdgcn_s_sleep(1);
        }
        __syncthreads();
        __threadfence();   // acquire: invalidate so new h is observed
    }
}

// ---------------------------------------------------------------------------
// Head: out[b] = dot(h_last[b,:], w_lin) + b_lin
// ---------------------------------------------------------------------------
__global__ __launch_bounds__(128) void final_linear(const float* __restrict__ hlast,
                                                    const float* __restrict__ wlin,
                                                    const float* __restrict__ blin,
                                                    float* __restrict__ out) {
    int b = blockIdx.x, tid = threadIdx.x;
    float s = 0.0f;
    for (int i = tid; i < H_; i += 128) s += hlast[(size_t)b * H_ + i] * wlin[i];
    __shared__ float red[128];
    red[tid] = s;
    __syncthreads();
    for (int o = 64; o > 0; o >>= 1) {
        if (tid < o) red[tid] += red[tid + o];
        __syncthreads();
    }
    if (tid == 0) out[b] = red[0] + blin[0];
}

// ---------------------------------------------------------------------------
extern "C" void kernel_launch(void* const* d_in, const int* in_sizes, int n_in,
                              void* d_out, int out_size, void* d_ws, size_t ws_size,
                              hipStream_t stream) {
    const float* x     = (const float*)d_in[0];
    const float* w_ih0 = (const float*)d_in[1];
    const float* w_hh0 = (const float*)d_in[2];
    const float* b_ih0 = (const float*)d_in[3];
    const float* b_hh0 = (const float*)d_in[4];
    const float* w_ih1 = (const float*)d_in[5];
    const float* w_hh1 = (const float*)d_in[6];
    const float* b_ih1 = (const float*)d_in[7];
    const float* b_hh1 = (const float*)d_in[8];
    const float* w_lin = (const float*)d_in[9];
    const float* b_lin = (const float*)d_in[10];
    float* out = (float*)d_out;

    char* wsb = (char*)d_ws;
    size_t off = 0;
    auto take = [&](size_t bytes) -> char* {
        char* p = wsb + off;
        off += (bytes + 255) & ~(size_t)255;
        return p;
    };

    __bf16* xb    = (__bf16*)take((size_t)B_ * T_ * D_ * sizeof(__bf16)); // 32 MB
    __bf16* wi0b  = (__bf16*)take((size_t)G_ * D_ * sizeof(__bf16));
    __bf16* wh0b  = (__bf16*)take((size_t)G_ * H_ * sizeof(__bf16));
    __bf16* wi1b  = (__bf16*)take((size_t)G_ * H_ * sizeof(__bf16));
    __bf16* wh1b  = (__bf16*)take((size_t)G_ * H_ * sizeof(__bf16));
    float*  bias0 = (float*)take((size_t)G_ * sizeof(float));
    float*  bias1 = (float*)take((size_t)G_ * sizeof(float));
    float*  xg    = (float*)take((size_t)B_ * T_ * G_ * sizeof(float));   // 256 MB
    __bf16* h0b   = (__bf16*)take((size_t)B_ * T_ * H_ * sizeof(__bf16)); // 32 MB
    __bf16* hbuf  = (__bf16*)take((size_t)2 * B_ * H_ * sizeof(__bf16));  // ping-pong
    float*  hlast = (float*)take((size_t)B_ * H_ * sizeof(float));
    unsigned int* bar = (unsigned int*)take(256);

    // ---- precision conversions -------------------------------------------
    cvt_f32_bf16<<<2048, 256, 0, stream>>>(x, xb, B_ * T_ * D_);
    cvt_f32_bf16<<<1024, 256, 0, stream>>>(w_ih0, wi0b, G_ * D_);
    cvt_f32_bf16<<<1024, 256, 0, stream>>>(w_hh0, wh0b, G_ * H_);
    cvt_f32_bf16<<<1024, 256, 0, stream>>>(w_ih1, wi1b, G_ * H_);
    cvt_f32_bf16<<<1024, 256, 0, stream>>>(w_hh1, wh1b, G_ * H_);
    bias_sum<<<8, 256, 0, stream>>>(b_ih0, b_hh0, bias0, G_);
    bias_sum<<<8, 256, 0, stream>>>(b_ih1, b_hh1, bias1, G_);

    const int M = B_ * T_;
    dim3 ggrid(M / 128, G_ / 64);

    // ---- layer 0 ----------------------------------------------------------
    gemm_bt<<<ggrid, 256, 0, stream>>>(xb, wi0b, xg, M, G_, D_);
    hipMemsetAsync(hbuf, 0, (size_t)2 * B_ * H_ * sizeof(__bf16), stream);
    hipMemsetAsync(bar, 0, 256, stream);
    lstm_scan<<<32, 256, 0, stream>>>(xg, wh0b, bias0, hbuf, h0b, nullptr, bar);

    // ---- layer 1 ----------------------------------------------------------
    gemm_bt<<<ggrid, 256, 0, stream>>>(h0b, wi1b, xg, M, G_, H_);
    hipMemsetAsync(hbuf, 0, (size_t)2 * B_ * H_ * sizeof(__bf16), stream);
    hipMemsetAsync(bar, 0, 256, stream);
    lstm_scan<<<32, 256, 0, stream>>>(xg, wh1b, bias1, hbuf, nullptr, hlast, bar);

    // ---- head -------------------------------------------------------------
    final_linear<<<B_, 128, 0, stream>>>(hlast, w_lin, b_lin, out);
}
